// _PixelCNN_62878321213758
// MI455X (gfx1250) — compile-verified, matrix operand order checked by reading, not run on hardware
//
#include <hip/hip_runtime.h>

typedef __attribute__((ext_vector_type(16))) _Float16 v16h;
typedef __attribute__((ext_vector_type(8)))  float    v8f;

#define F     160
#define HDIM  64
#define WDIM  64
#define BATCH 32

// Reorder a channel index within a 32-chunk so that lane-group 0 (lanes 0-15)
// reads halfs [0..15] = K {0..7, 16..23} and lane-group 1 reads halfs [16..31]
// = K {8..15, 24..31}, matching the WMMA 16-bit B-operand layout.
__device__ __forceinline__ int chpos(int kl) {
  return (kl & 7) + ((kl & 8) ? 16 : 0) + ((kl & 16) ? 8 : 0);
}

// ---------------- weight packing: 3x3 mask-B convs -------------------------
// wp layout per conv: [ocTile(10)][tap(5)][kchunk(5)][lane(32)][half(16)]
__global__ void pack_w3_kernel(const float* __restrict__ wres,
                               _Float16* __restrict__ wp) {
  int idx = blockIdx.x * 256 + threadIdx.x;           // 6*128000 total
  int hh   = idx & 15;
  int lane = (idx >> 4) & 31;
  int rest = idx >> 9;
  int kc = rest % 5; rest /= 5;
  int tap = rest % 5; rest /= 5;
  int m = rest % 10;
  int c = rest / 10;                                   // conv index 0..5
  const int kyA[5] = {0,0,0,1,1};                      // mask-B taps
  const int kxA[5] = {0,1,2,0,1};
  int g  = lane >> 4;
  int kl = (hh & 7) + ((hh >> 3) << 4) + (g << 3);     // A-operand K interleave
  int oc = m * 16 + (lane & 15);
  int ic = kc * 32 + kl;
  float wv = wres[(((c * F + oc) * F + ic) * 3 + kyA[tap]) * 3 + kxA[tap]];
  wp[idx] = (_Float16)wv;
}

// ---------------- weight packing: 1x1 convs --------------------------------
__global__ void pack_w1x1_kernel(const float* __restrict__ w,
                                 _Float16* __restrict__ wp, int cout) {
  int idx = blockIdx.x * 256 + threadIdx.x;           // nOcTiles*5*512 total
  int hh   = idx & 15;
  int lane = (idx >> 4) & 31;
  int rest = idx >> 9;
  int kc = rest % 5;
  int m  = rest / 5;
  int g  = lane >> 4;
  int kl = (hh & 7) + ((hh >> 3) << 4) + (g << 3);
  int oc = m * 16 + (lane & 15);
  int ic = kc * 32 + kl;
  float wv = (oc < cout) ? w[oc * F + ic] : 0.0f;
  wp[idx] = (_Float16)wv;
}

// ---------------- first layer: 7x7 mask-A conv, C_in=3 (scalar) ------------
__global__ void conv7x7_maskA_kernel(const float* __restrict__ x,
                                     const float* __restrict__ w0,
                                     const float* __restrict__ b0,
                                     float* __restrict__ h) {
  int idx = blockIdx.x * 256 + threadIdx.x;           // 32*160*64*64 total
  int xw = idx & 63;
  int yw = (idx >> 6) & 63;
  int p  = idx >> 12;
  int oc = p % F;
  int b  = p / F;
  float acc = b0[oc];
  for (int ky = 0; ky <= 3; ++ky) {                   // mask A: rows 0-2 full, row 3 cols 0-2
    int kxmax = (ky < 3) ? 7 : 3;
    int yy = yw + ky - 3;
    if (yy < 0) continue;
    for (int kx = 0; kx < kxmax; ++kx) {
      int xx = xw + kx - 3;
      if (xx < 0 || xx >= WDIM) continue;
      for (int ci = 0; ci < 3; ++ci) {
        float v = x[((b * 3 + ci) * HDIM + yy) * WDIM + xx] * 2.0f - 1.0f;
        acc += v * w0[((oc * 3 + ci) * 7 + ky) * 7 + kx];
      }
    }
  }
  h[idx] = acc;
}

// ---------------- 3x3 mask-B conv as implicit GEMM via WMMA ----------------
// out = conv3x3_maskB(relu(in)) + bias (+ skip)
// Each wave owns one 16-px xblock and 5 ocTiles; B fragments loaded once and
// reused across the 5 accumulators (5x less LDS read traffic).
__global__ void __launch_bounds__(256)
conv3x3_wmma_kernel(const float* __restrict__ in, const _Float16* __restrict__ wp,
                    const float* __restrict__ bias, const float* __restrict__ skip,
                    float* __restrict__ out) {
  __shared__ v16h lds[2 * 66 * 11];                   // 2 rows x 66 px x 352B
  _Float16* ldsH = (_Float16*)lds;
  const int bh   = blockIdx.x;                        // b*64 + y
  const int bidx = bh >> 6;
  const int y    = bh & 63;
  const int tid  = threadIdx.x;

  // Stage rows y-1, y (+/-1 x halo), relu + cvt f16, channel-reordered.
  // Pixel-fastest iteration => lanes read contiguous global addresses.
  for (int i = tid; i < 2 * 66 * F; i += 256) {
    int px  = i % 66;
    int rem = i / 66;
    int ic  = rem % F;
    int r   = rem / F;
    int gx = px - 1;
    int gy = y - 1 + r;
    float v = 0.0f;
    if (gx >= 0 && gx < WDIM && gy >= 0) {
      v = in[((bidx * F + ic) * HDIM + gy) * WDIM + gx];
      v = v > 0.0f ? v : 0.0f;
    }
    ldsH[(r * 66 + px) * 176 + (ic >> 5) * 32 + chpos(ic & 31)] = (_Float16)v;
  }
  __syncthreads();

  const int wave = tid >> 5;
  const int lane = tid & 31;
  const int g    = lane >> 4;
  const int col  = lane & 15;
  const int x0   = (wave & 3) * 16;                   // wave-private xblock
  const int ocT0 = wave >> 2;                         // 0 or 1
  const int rA[5]  = {0, 0, 0, 1, 1};
  const int dxA[5] = {-1, 0, 1, -1, 0};
  const v16h* wpv = (const v16h*)wp;

  v8f acc[5];
#pragma unroll
  for (int jj = 0; jj < 5; ++jj) {
    int ocT = ocT0 + 2 * jj;
#pragma unroll
    for (int e = 0; e < 8; ++e) acc[jj][e] = bias[ocT * 16 + e + 8 * g];
  }

#pragma unroll
  for (int t = 0; t < 5; ++t) {
    int px = x0 + col + dxA[t] + 1;
    int pb = (rA[t] * 66 + px) * 11;
#pragma unroll
    for (int kc = 0; kc < 5; ++kc) {
      v16h bb = lds[pb + kc * 2 + g];                 // loaded once, used 5x
#pragma unroll
      for (int jj = 0; jj < 5; ++jj) {
        int ocT = ocT0 + 2 * jj;
        v16h a = wpv[((ocT * 5 + t) * 5 + kc) * 32 + lane];
        acc[jj] = __builtin_amdgcn_wmma_f32_16x16x32_f16(
            false, a, false, bb, (short)0, acc[jj], false, false);
      }
    }
  }

#pragma unroll
  for (int jj = 0; jj < 5; ++jj) {
    int ocT = ocT0 + 2 * jj;
#pragma unroll
    for (int e = 0; e < 8; ++e) {
      int oc = ocT * 16 + e + 8 * g;
      int o  = ((bidx * F + oc) * HDIM + y) * WDIM + x0 + col;
      float val = acc[jj][e];
      if (skip) val += skip[o];
      out[o] = val;
    }
  }
}

// ---------------- 1x1 conv (relu fused) via WMMA ---------------------------
template <int NOCT>
__global__ void __launch_bounds__(256)
conv1x1_wmma_kernel(const float* __restrict__ in, const _Float16* __restrict__ wp,
                    const float* __restrict__ bias, float* __restrict__ out,
                    int cout) {
  __shared__ v16h lds[64 * 11];
  _Float16* ldsH = (_Float16*)lds;
  const int bh   = blockIdx.x;
  const int bidx = bh >> 6;
  const int y    = bh & 63;
  const int tid  = threadIdx.x;

  for (int i = tid; i < 64 * F; i += 256) {           // px-fastest: coalesced
    int px = i & 63;
    int ic = i >> 6;
    float v = in[((bidx * F + ic) * HDIM + y) * WDIM + px];
    v = v > 0.0f ? v : 0.0f;
    ldsH[px * 176 + (ic >> 5) * 32 + chpos(ic & 31)] = (_Float16)v;
  }
  __syncthreads();

  const int wave = tid >> 5;
  const int lane = tid & 31;
  const int g    = lane >> 4;
  const int col  = lane & 15;
  const int x0   = (wave & 3) * 16;
  const int ocT0 = wave >> 2;
  const int NJ   = NOCT / 2;                          // tiles per wave
  const v16h* wpv = (const v16h*)wp;

  v8f acc[NJ];
#pragma unroll
  for (int jj = 0; jj < NJ; ++jj) {
    int ocT = ocT0 + 2 * jj;
#pragma unroll
    for (int e = 0; e < 8; ++e) {
      int oc = ocT * 16 + e + 8 * g;
      acc[jj][e] = (oc < cout) ? bias[oc] : 0.0f;
    }
  }

  int pb = (x0 + col) * 11;
#pragma unroll
  for (int kc = 0; kc < 5; ++kc) {
    v16h bb = lds[pb + kc * 2 + g];
#pragma unroll
    for (int jj = 0; jj < NJ; ++jj) {
      int ocT = ocT0 + 2 * jj;
      v16h a = wpv[(ocT * 5 + kc) * 32 + lane];
      acc[jj] = __builtin_amdgcn_wmma_f32_16x16x32_f16(
          false, a, false, bb, (short)0, acc[jj], false, false);
    }
  }

#pragma unroll
  for (int jj = 0; jj < NJ; ++jj) {
    int ocT = ocT0 + 2 * jj;
#pragma unroll
    for (int e = 0; e < 8; ++e) {
      int oc = ocT * 16 + e + 8 * g;
      if (oc < cout) {
        out[((bidx * cout + oc) * HDIM + y) * WDIM + x0 + col] = acc[jj][e];
      }
    }
  }
}

extern "C" void kernel_launch(void* const* d_in, const int* in_sizes, int n_in,
                              void* d_out, int out_size, void* d_ws, size_t ws_size,
                              hipStream_t stream) {
  const float* samples = (const float*)d_in[0];
  const float* w0      = (const float*)d_in[1];
  const float* b0      = (const float*)d_in[2];
  const float* w_res   = (const float*)d_in[3];
  const float* b_res   = (const float*)d_in[4];
  const float* w_out1  = (const float*)d_in[5];
  const float* b_out1  = (const float*)d_in[6];
  const float* w_out2  = (const float*)d_in[7];
  const float* b_out2  = (const float*)d_in[8];
  float* out = (float*)d_out;

  char* ws = (char*)d_ws;
  const size_t szAct = (size_t)BATCH * F * HDIM * WDIM * sizeof(float); // 83.9 MB
  float*    h   = (float*)ws;
  float*    t   = (float*)(ws + szAct);
  _Float16* wp3 = (_Float16*)(ws + 2 * szAct);       // 6 x 128000 halfs
  _Float16* wp1 = wp3 + 6 * 128000;                  // 25600 halfs
  _Float16* wp2 = wp1 + 25600;                       // 5120 halfs

  // Pack weights into WMMA operand layout (tiny; cached in L2 afterwards).
  pack_w3_kernel<<<3000, 256, 0, stream>>>(w_res, wp3);
  pack_w1x1_kernel<<<100, 256, 0, stream>>>(w_out1, wp1, F);
  pack_w1x1_kernel<<<20, 256, 0, stream>>>(w_out2, wp2, 30);

  // First layer (scalar; <2% of FLOPs).
  conv7x7_maskA_kernel<<<(BATCH * F * HDIM * WDIM) / 256, 256, 0, stream>>>(
      samples, w0, b0, h);

  // Residual blocks: t = conv(relu(h)); h = h + conv(relu(t)).
  for (int i = 0; i < 3; ++i) {
    conv3x3_wmma_kernel<<<BATCH * HDIM, 256, 0, stream>>>(
        h, wp3 + (2 * i) * 128000, b_res + (2 * i) * F, nullptr, t);
    conv3x3_wmma_kernel<<<BATCH * HDIM, 256, 0, stream>>>(
        t, wp3 + (2 * i + 1) * 128000, b_res + (2 * i + 1) * F, h, h);
  }

  // 1x1 heads: h1 = conv(relu(h)); logits = conv(relu(h1)).
  conv1x1_wmma_kernel<10><<<BATCH * HDIM, 256, 0, stream>>>(h, wp1, b_out1, t, F);
  conv1x1_wmma_kernel<2><<<BATCH * HDIM, 256, 0, stream>>>(t, wp2, b_out2, out, 30);
}
